// WindowAttention_35966056137224
// MI455X (gfx1250) — matrix operand order repaired and checked by reading, not compiled, not run on hardware
//
#include <hip/hip_runtime.h>

// ---------------- CDNA5 WMMA types ----------------
typedef __attribute__((ext_vector_type(16))) __bf16 v16bf;
typedef __attribute__((ext_vector_type(8)))  float  v8f;

#define N_TOK 64
#define DIM   192
#define HEADS 6
#define DH    32
#define SCALE 0.17677669529663687f   // 1/sqrt(32)

// LDS strides (elements). bf16 strides chosen so rows are 16B-aligned.
#define XS  200   // x / O buffer stride (row = 400 B)
#define QKS 392   // q|k buffer stride   (row = 784 B)
#define VTS 72    // v-transposed stride (row = 144 B)
#define SS  65    // scores stride (f32) -> conflict-free softmax rows
#define PS  72    // probs stride (bf16, row = 144 B)

__device__ __forceinline__ unsigned short f2bf(float x) {
  unsigned int u = __float_as_uint(x);
  u += 0x7FFFu + ((u >> 16) & 1u);       // round-to-nearest-even
  return (unsigned short)(u >> 16);
}

__device__ __forceinline__ v8f zero8() {
  v8f z = {0.f, 0.f, 0.f, 0.f, 0.f, 0.f, 0.f, 0.f};
  return z;
}

union FragU { v16bf v; uint4 q[2]; };

// A-matrix fragment (16x32 bf16): lane l<16 -> row l, K c..c+15;
// lane l>=16 -> row l-16, K c+16..c+31.  Two 16B loads.
__device__ __forceinline__ v16bf load_fragA(const unsigned short* base, int row0,
                                            int stride, int col0, int nn, int hi) {
  const unsigned short* p = base + (row0 + nn) * stride + col0 + hi * 16;
  FragU f;
  f.q[0] = *(const uint4*)(p);
  f.q[1] = *(const uint4*)(p + 8);
  return f.v;
}

// B-matrix fragment (32x16 bf16), source stored column-major-as-rows
// (memory row n == matrix column n): lane l<16 -> col l, K 0..15;
// lane l>=16 -> col l-16, K 16..31.  Two 16B loads.
__device__ __forceinline__ v16bf load_fragB(const unsigned short* base, int n0,
                                            int stride, int k0, int nn, int hi) {
  const unsigned short* p = base + (n0 + nn) * stride + k0 + hi * 16;
  FragU f;
  f.q[0] = *(const uint4*)(p);
  f.q[1] = *(const uint4*)(p + 8);
  return f.v;
}

__device__ __forceinline__ v8f bf16_wmma(v16bf a, v16bf b, v8f c) {
  return __builtin_amdgcn_wmma_f32_16x16x32_bf16(false, a, false, b,
                                                 (short)0, c, false, false);
}

// C-tile (16x16 f32) -> bf16 LDS, (acc+bias)*mul.  VGPR r: row m0+r+8*hi, col nn.
__device__ __forceinline__ void store_qk(unsigned short* dst, int m0, int col,
                                         int stride, const v8f& acc, float bias,
                                         float mul, int nn, int hi) {
#pragma unroll
  for (int r = 0; r < 8; ++r) {
    int m = m0 + r + hi * 8;
    dst[m * stride + col + nn] = f2bf((acc[r] + bias) * mul);
  }
}

// C-tile stored transposed into v^T buffer: row = head-dim col, col = token.
__device__ __forceinline__ void store_vT(unsigned short* dst, int m0, int dcol0,
                                         const v8f& acc, float bias, int nn, int hi) {
  int dcol = dcol0 + nn;
#pragma unroll
  for (int r = 0; r < 8; ++r) {
    int tok = m0 + r + hi * 8;
    dst[dcol * VTS + tok] = f2bf(acc[r] + bias);
  }
}

// ---------------- prep: weights->bf16, bias table gather ----------------
__global__ void prep_kernel(const float* __restrict__ qkv_w,
                            const float* __restrict__ proj_w,
                            const float* __restrict__ rpb,
                            const int* __restrict__ rel_idx,
                            unsigned short* __restrict__ wqkv,
                            unsigned short* __restrict__ wproj,
                            float* __restrict__ abias) {
  int i = blockIdx.x * blockDim.x + threadIdx.x;
  if (i < 576 * 192) wqkv[i] = f2bf(qkv_w[i]);
  if (i < 192 * 192) wproj[i] = f2bf(proj_w[i]);
  if (i < 64 * 64) {
    int rel = rel_idx[i];
#pragma unroll
    for (int h = 0; h < HEADS; ++h) abias[h * 4096 + i] = rpb[rel * HEADS + h];
  }
}

// ---------------- fused window attention: one block per window ----------------
__global__ __launch_bounds__(256) void winattn_kernel(
    const float* __restrict__ x, const float* __restrict__ qkv_b,
    const float* __restrict__ proj_b, const unsigned short* __restrict__ wqkv,
    const unsigned short* __restrict__ wproj, const float* __restrict__ abias,
    float* __restrict__ out) {
  // 129,280 B of LDS (fits CDNA5's 320 KB/WGP; 2 blocks resident per WGP)
  __shared__ __align__(16) unsigned short s_xo[N_TOK * XS];   // x, then O (reused)
  __shared__ __align__(16) unsigned short s_qk[N_TOK * QKS];  // q | k  (bf16)
  __shared__ __align__(16) unsigned short s_vt[DIM * VTS];    // v^T    (bf16)
  __shared__ __align__(16) float          s_S[N_TOK * SS];    // scores (f32)
  __shared__ __align__(16) unsigned short s_P[N_TOK * PS];    // probs  (bf16)

  const int tid  = threadIdx.x;
  const int lane = tid & 31;
  const int wave = tid >> 5;
  const int nn   = lane & 15;
  const int hi   = lane >> 4;
  const int win  = blockIdx.x;

  // ---- Phase 1: x -> bf16 LDS (coalesced global reads) ----
  const float* xg = x + (size_t)win * (N_TOK * DIM);
  for (int i = tid; i < N_TOK * DIM; i += 256)
    s_xo[(i / DIM) * XS + (i % DIM)] = f2bf(xg[i]);
  __syncthreads();

  // ---- Phase 2: qkv = x @ W^T + b   (4x36 tiles, K=192) ----
  // Each wave owns column-tiles nt, reuses one B fragment across 4 row-tiles.
  for (int nt = wave; nt < 36; nt += 8) {
    const int n0 = nt * 16;
    v8f c0 = zero8(), c1 = zero8(), c2 = zero8(), c3 = zero8();
#pragma unroll
    for (int kb = 0; kb < 6; ++kb) {
      v16bf b = load_fragB(wqkv, n0, DIM, kb * 32, nn, hi);
      c0 = bf16_wmma(load_fragA(s_xo,  0, XS, kb * 32, nn, hi), b, c0);
      c1 = bf16_wmma(load_fragA(s_xo, 16, XS, kb * 32, nn, hi), b, c1);
      c2 = bf16_wmma(load_fragA(s_xo, 32, XS, kb * 32, nn, hi), b, c2);
      c3 = bf16_wmma(load_fragA(s_xo, 48, XS, kb * 32, nn, hi), b, c3);
    }
    const float bias = qkv_b[n0 + nn];
    if (n0 < DIM) {                      // q: pre-scale by 1/sqrt(d)
      store_qk(s_qk,  0, n0, QKS, c0, bias, SCALE, nn, hi);
      store_qk(s_qk, 16, n0, QKS, c1, bias, SCALE, nn, hi);
      store_qk(s_qk, 32, n0, QKS, c2, bias, SCALE, nn, hi);
      store_qk(s_qk, 48, n0, QKS, c3, bias, SCALE, nn, hi);
    } else if (n0 < 2 * DIM) {           // k: row-major (rows become B cols)
      store_qk(s_qk,  0, n0, QKS, c0, bias, 1.0f, nn, hi);
      store_qk(s_qk, 16, n0, QKS, c1, bias, 1.0f, nn, hi);
      store_qk(s_qk, 32, n0, QKS, c2, bias, 1.0f, nn, hi);
      store_qk(s_qk, 48, n0, QKS, c3, bias, 1.0f, nn, hi);
    } else {                             // v: store transposed for PV B-frags
      store_vT(s_vt,  0, n0 - 2 * DIM, c0, bias, nn, hi);
      store_vT(s_vt, 16, n0 - 2 * DIM, c1, bias, nn, hi);
      store_vT(s_vt, 32, n0 - 2 * DIM, c2, bias, nn, hi);
      store_vT(s_vt, 48, n0 - 2 * DIM, c3, bias, nn, hi);
    }
  }
  __syncthreads();

  // ---- Phase 3: per-head attention ----
  for (int h = 0; h < HEADS; ++h) {
    // S = q_h @ k_h^T + bias  (16 tiles, K=32 -> one WMMA each; 2 tiles/wave)
#pragma unroll
    for (int i = 0; i < 2; ++i) {
      const int t  = wave * 2 + i;
      const int mt = t >> 2, nt = t & 3;
      v16bf a = load_fragA(s_qk, mt * 16, QKS, h * DH, nn, hi);
      v16bf b = load_fragB(s_qk, nt * 16, QKS, DIM + h * DH, nn, hi);
      v8f acc = bf16_wmma(a, b, zero8());
      const float* bp = abias + h * 4096 + nt * 16 + nn;
#pragma unroll
      for (int r = 0; r < 8; ++r) {
        int m = mt * 16 + r + hi * 8;
        s_S[m * SS + nt * 16 + nn] = acc[r] + bp[m * 64];
      }
    }
    __syncthreads();

    // row softmax: 64 threads, one full row each (stride 65 -> conflict-free)
    if (tid < N_TOK) {
      float* row = s_S + tid * SS;
      float mx = -1e30f;
#pragma unroll 4
      for (int j = 0; j < 64; ++j) mx = fmaxf(mx, row[j]);
      float sum = 0.f;
#pragma unroll 4
      for (int j = 0; j < 64; ++j) { float e = __expf(row[j] - mx); row[j] = e; sum += e; }
      const float inv = 1.0f / sum;
      unsigned short* pr = s_P + tid * PS;
#pragma unroll 4
      for (int j = 0; j < 64; ++j) pr[j] = f2bf(row[j] * inv);
    }
    __syncthreads();

    // O_h = P @ V_h  (4x2 tiles, K=64 -> one tile per wave)
    {
      const int mt = wave >> 1, nt = wave & 1;
      v8f acc = zero8();
#pragma unroll
      for (int kb = 0; kb < 2; ++kb) {
        v16bf a = load_fragA(s_P, mt * 16, PS, kb * 32, nn, hi);
        v16bf b = load_fragB(s_vt, h * DH + nt * 16, VTS, kb * 32, nn, hi);
        acc = bf16_wmma(a, b, acc);
      }
#pragma unroll
      for (int r = 0; r < 8; ++r) {
        int m = mt * 16 + r + hi * 8;
        s_xo[m * XS + h * DH + nt * 16 + nn] = f2bf(acc[r]);  // O reuses x LDS
      }
    }
    __syncthreads();
  }

  // ---- Phase 4: out = O @ proj_w^T + b  (4x12 tiles, K=192) ----
  float* og = out + (size_t)win * (N_TOK * DIM);
  for (int nt = wave; nt < 12; nt += 8) {
    const int n0 = nt * 16;
    v8f c0 = zero8(), c1 = zero8(), c2 = zero8(), c3 = zero8();
#pragma unroll
    for (int kb = 0; kb < 6; ++kb) {
      v16bf b = load_fragB(wproj, n0, DIM, kb * 32, nn, hi);
      c0 = bf16_wmma(load_fragA(s_xo,  0, XS, kb * 32, nn, hi), b, c0);
      c1 = bf16_wmma(load_fragA(s_xo, 16, XS, kb * 32, nn, hi), b, c1);
      c2 = bf16_wmma(load_fragA(s_xo, 32, XS, kb * 32, nn, hi), b, c2);
      c3 = bf16_wmma(load_fragA(s_xo, 48, XS, kb * 32, nn, hi), b, c3);
    }
    const float bias = proj_b[n0 + nn];
#pragma unroll
    for (int r = 0; r < 8; ++r) {
      og[(0  + r + hi * 8) * DIM + n0 + nn] = c0[r] + bias;
      og[(16 + r + hi * 8) * DIM + n0 + nn] = c1[r] + bias;
      og[(32 + r + hi * 8) * DIM + n0 + nn] = c2[r] + bias;
      og[(48 + r + hi * 8) * DIM + n0 + nn] = c3[r] + bias;
    }
  }
}

// ---------------- host launcher ----------------
extern "C" void kernel_launch(void* const* d_in, const int* in_sizes, int n_in,
                              void* d_out, int out_size, void* d_ws, size_t ws_size,
                              hipStream_t stream) {
  const float* x      = (const float*)d_in[0];
  const float* qkv_w  = (const float*)d_in[1];
  const float* qkv_b  = (const float*)d_in[2];
  const float* proj_w = (const float*)d_in[3];
  const float* proj_b = (const float*)d_in[4];
  const float* rpb    = (const float*)d_in[5];
  const int*   relidx = (const int*)d_in[6];
  float* out = (float*)d_out;

  // workspace carve-up (16B-aligned regions)
  char* ws = (char*)d_ws;
  unsigned short* wqkv  = (unsigned short*)(ws);              // 576*192*2 = 221184 B
  unsigned short* wproj = (unsigned short*)(ws + 221184);     // 192*192*2 =  73728 B
  float*          abias = (float*)(ws + 294912);              // 6*64*64*4 =  98304 B

  prep_kernel<<<(576 * 192 + 255) / 256, 256, 0, stream>>>(
      qkv_w, proj_w, rpb, relidx, wqkv, wproj, abias);

  winattn_kernel<<<4096, 256, 0, stream>>>(
      x, qkv_b, proj_b, wqkv, wproj, abias, out);
}
